// MemoryUnit_74268574482523
// MI455X (gfx1250) — matrix operand order repaired and checked by reading, not compiled, not run on hardware
//
#include <hip/hip_runtime.h>
#include <hip/hip_bf16.h>
#include <math.h>

// Problem constants
#define BB   512
#define HH   1024
#define MM   131072
#define DD   128
#define EPS  1e-8f
#define INV_B (1.0f/512.0f)

typedef __bf16 bf16;
typedef __attribute__((ext_vector_type(16))) __bf16 v16bf;
typedef __attribute__((ext_vector_type(8)))  __bf16 v8bf;
typedef __attribute__((ext_vector_type(8)))  float  v8f;

union AV { v16bf v; v8bf h[2]; };

static __device__ inline v8f wmma_bf16(v16bf a, v16bf b, v8f c) {
  return __builtin_amdgcn_wmma_f32_16x16x32_bf16(
      false, a, false, b, (short)0, c, false, false);
}

// ---------------------------------------------------------------------------
// K1: projections. One block per batch row b.
// Computes read_q, write_q (scaled into bf16), key strength, eraseT/addT bf16.
// ---------------------------------------------------------------------------
__global__ __launch_bounds__(256)
void k_proj(const float* __restrict__ h,
            const float* __restrict__ Wr, const float* __restrict__ br,
            const float* __restrict__ Ww, const float* __restrict__ bw,
            const float* __restrict__ Wk, const float* __restrict__ bk,
            const float* __restrict__ We, const float* __restrict__ be,
            const float* __restrict__ Wa, const float* __restrict__ ba,
            bf16* __restrict__ qr, bf16* __restrict__ qw,
            bf16* __restrict__ eT, bf16* __restrict__ aT)
{
  __shared__ float sh[HH];
  __shared__ float outbuf[512];
  __shared__ float red[256];
  __shared__ float s_ks;

  const int b = blockIdx.x;
  const int t = threadIdx.x;

  for (int i = t; i < HH; i += 256) sh[i] = h[(size_t)b * HH + i];
  __syncthreads();

  // key strength: softplus(h . Wk + bk)
  float kp = 0.f;
  for (int i = t; i < HH; i += 256) kp += sh[i] * Wk[i];
  red[t] = kp;
  __syncthreads();
  for (int s = 128; s > 0; s >>= 1) {
    if (t < s) red[t] += red[t + s];
    __syncthreads();
  }
  if (t == 0) {
    float kl = red[0] + bk[0];
    s_ks = (kl > 20.f) ? kl : log1pf(__expf(kl));
  }
  __syncthreads();

  // 4 linear layers: 512 output dims total, 2 per thread
  for (int j = t; j < 512; j += 256) {
    const int mat = j >> 7, d = j & 127;
    const float* W = (mat == 0) ? Wr : (mat == 1) ? Ww : (mat == 2) ? We : Wa;
    const float* bias = (mat == 0) ? br : (mat == 1) ? bw : (mat == 2) ? be : ba;
    const float4* r4 = (const float4*)(W + (size_t)d * HH);
    float s = 0.f;
#pragma unroll 4
    for (int k = 0; k < HH / 4; ++k) {
      float4 w4 = r4[k];
      s += w4.x * sh[4 * k + 0] + w4.y * sh[4 * k + 1]
         + w4.z * sh[4 * k + 2] + w4.w * sh[4 * k + 3];
    }
    outbuf[j] = s + bias[d];
  }
  __syncthreads();

  // norms of read_q (outbuf[0..127]) and write_q (outbuf[128..255])
  red[t] = outbuf[t] * outbuf[t];
  __syncthreads();
  for (int s = 64; s > 0; s >>= 1) {
    if ((t & 127) < s) red[t] += red[t + s];
    __syncthreads();
  }
  const float snr = s_ks / fmaxf(sqrtf(red[0]), EPS);
  const float snw = 1.0f / fmaxf(sqrtf(red[128]), EPS);

  for (int j = t; j < 512; j += 256) {
    const int mat = j >> 7, d = j & 127;
    const float v = outbuf[j];
    if (mat == 0)      qr[(size_t)b * DD + d] = (bf16)(v * snr);
    else if (mat == 1) qw[(size_t)b * DD + d] = (bf16)(v * snw);
    else if (mat == 2) eT[(size_t)d * BB + b] = (bf16)(1.0f / (1.0f + __expf(-v)));
    else               aT[(size_t)d * BB + b] = (bf16)tanhf(v);
  }
}

// ---------------------------------------------------------------------------
// K2: memory prep. 64 rows per block.
//   mn[m][d]  = bf16(mem[m][d] / max(|mem[m]|,eps))   (normalized rows)
//   mT[d][m]  = bf16(mem[m][d])                        (transpose, raw)
// ---------------------------------------------------------------------------
__global__ __launch_bounds__(256)
void k_memprep(const float* __restrict__ mem,
               bf16* __restrict__ mn, bf16* __restrict__ mT)
{
  __shared__ float tile[64][DD + 1];
  __shared__ float invn[64];
  const int m0 = blockIdx.x * 64;
  const int t = threadIdx.x;

  for (int idx = t; idx < 64 * DD; idx += 256) {
    const int r = idx >> 7, c = idx & 127;
    tile[r][c] = mem[(size_t)(m0 + r) * DD + c];
  }
  __syncthreads();
  if (t < 64) {
    float s = 0.f;
#pragma unroll 4
    for (int c = 0; c < DD; ++c) { float v = tile[t][c]; s += v * v; }
    invn[t] = 1.0f / fmaxf(sqrtf(s), EPS);
  }
  __syncthreads();
  for (int idx = t; idx < 64 * DD; idx += 256) {
    const int r = idx >> 7, c = idx & 127;
    mn[(size_t)(m0 + r) * DD + c] = (bf16)(tile[r][c] * invn[r]);
  }
  for (int idx = t; idx < 64 * DD; idx += 256) {
    const int d = idx >> 6, r = idx & 63;
    mT[(size_t)d * MM + m0 + r] = (bf16)tile[r][d];
  }
}

// ---------------------------------------------------------------------------
// K3: fused cosine attention. grid = (64 m-chunks, 8 b-tiles), 8 waves.
// Block: 64 queries x 2048 memory rows. Per 32-m step:
//   S_read/S_write via WMMA bf16 (normalized mem, scaled queries),
//   p = exp(S); p_write -> d_out (unnormalized); Z partials; p_read staged
//   through LDS into A layout and WMMA'd against memT for read-vector partial.
// ---------------------------------------------------------------------------
__global__ __launch_bounds__(256)
void k_attn(const bf16* __restrict__ qr, const bf16* __restrict__ qw,
            const bf16* __restrict__ mn, const bf16* __restrict__ mT,
            float* __restrict__ ww, float* __restrict__ Zr_part,
            float* __restrict__ Zw_part, float* __restrict__ rv_part)
{
  __shared__ bf16 pb[64][40];        // p_read tile, [b][m], padded
  __shared__ float zrr[8][16];
  __shared__ float zrw[8][16];

  const int chunk = blockIdx.x;      // 0..63, 2048 mem rows each
  const int bt = blockIdx.y;         // 0..7
  const int b0 = bt * 64;
  const int m_begin = chunk * 2048;

  const int t = threadIdx.x;
  const int w = t >> 5, lane = t & 31;
  const int wrow = w >> 1, wcol = w & 1;
  const int lh = lane >> 4, ln = lane & 15;

  // A operands (queries), resident for the whole block
  AV aR[4], aW[4];
  {
    const bf16* prr = qr + (size_t)(b0 + wrow * 16 + ln) * DD;
    const bf16* pww = qw + (size_t)(b0 + wrow * 16 + ln) * DD;
#pragma unroll
    for (int c = 0; c < 4; ++c) {
      aR[c].h[0] = *(const v8bf*)(prr + c * 32 + lh * 16);
      aR[c].h[1] = *(const v8bf*)(prr + c * 32 + lh * 16 + 8);
      aW[c].h[0] = *(const v8bf*)(pww + c * 32 + lh * 16);
      aW[c].h[1] = *(const v8bf*)(pww + c * 32 + lh * 16 + 8);
    }
  }

  float zr8[8], zw8[8];
#pragma unroll
  for (int r = 0; r < 8; ++r) { zr8[r] = 0.f; zw8[r] = 0.f; }
  v8f rvacc[4];
#pragma unroll
  for (int n = 0; n < 4; ++n) {
    v8f z = {0.f, 0.f, 0.f, 0.f, 0.f, 0.f, 0.f, 0.f};
    rvacc[n] = z;
  }

  for (int mt = m_begin; mt < m_begin + 2048; mt += 32) {
    const int mcol = mt + wcol * 16 + ln;

    // B operand: normalized memory rows (S = Q @ Mem^T)
    v16bf bm[4];
#pragma unroll
    for (int c = 0; c < 4; ++c)
      bm[c] = *(const v16bf*)(mn + (size_t)mcol * DD + c * 32 + lh * 16);

    v8f sr = {0.f, 0.f, 0.f, 0.f, 0.f, 0.f, 0.f, 0.f};
    v8f sw = {0.f, 0.f, 0.f, 0.f, 0.f, 0.f, 0.f, 0.f};
#pragma unroll
    for (int c = 0; c < 4; ++c) {
      sr = wmma_bf16(aR[c].v, bm[c], sr);
      sw = wmma_bf16(aW[c].v, bm[c], sw);
    }

    // exp (logits are bounded: no max subtraction needed)
#pragma unroll
    for (int r = 0; r < 8; ++r) {
      const float pr = __expf(sr[r]);
      const float pw = __expf(sw[r]);
      zr8[r] += pr; zw8[r] += pw;
      const int brow = b0 + wrow * 16 + r + 8 * lh;
      ww[(size_t)brow * MM + mcol] = pw;                 // unnormalized
      pb[wrow * 16 + r + 8 * lh][wcol * 16 + ln] = (bf16)pr;
    }
    __syncthreads();

    // read-vector partial: rv += p_read @ mem  (A from LDS, B from memT)
    AV ap;
    {
      const bf16* prow = &pb[wrow * 16 + ln][0];
      ap.h[0] = *(const v8bf*)(prow + lh * 16);
      ap.h[1] = *(const v8bf*)(prow + lh * 16 + 8);
    }
#pragma unroll
    for (int n = 0; n < 4; ++n) {
      const int dcol = wcol * 64 + n * 16 + ln;
      v16bf bv = *(const v16bf*)(mT + (size_t)dcol * MM + mt + lh * 16);
      rvacc[n] = wmma_bf16(ap.v, bv, rvacc[n]);
    }
    __syncthreads();
  }

  // reduce Z over the 16 columns held by this lane group
#pragma unroll
  for (int r = 0; r < 8; ++r) {
#pragma unroll
    for (int msk = 1; msk < 16; msk <<= 1) {
      zr8[r] += __shfl_xor(zr8[r], msk, 32);
      zw8[r] += __shfl_xor(zw8[r], msk, 32);
    }
  }
  if (ln == 0) {
#pragma unroll
    for (int r = 0; r < 8; ++r) {
      zrr[w][r + 8 * lh] = zr8[r];
      zrw[w][r + 8 * lh] = zw8[r];
    }
  }
  __syncthreads();
  if (t < 64) {
    const int wr2 = t >> 4, ri = t & 15;
    Zr_part[(size_t)chunk * BB + b0 + t] = zrr[wr2 * 2][ri] + zrr[wr2 * 2 + 1][ri];
    Zw_part[(size_t)chunk * BB + b0 + t] = zrw[wr2 * 2][ri] + zrw[wr2 * 2 + 1][ri];
  }

  // read-vector partials
#pragma unroll
  for (int n = 0; n < 4; ++n) {
#pragma unroll
    for (int r = 0; r < 8; ++r) {
      const int brow = b0 + wrow * 16 + r + 8 * lh;
      const int d = wcol * 64 + n * 16 + ln;
      rv_part[((size_t)chunk * BB + brow) * DD + d] = rvacc[n][r];
    }
  }
}

// ---------------------------------------------------------------------------
// K4: reduce Z partials + read-vector partials. grid = B blocks of 128.
// ---------------------------------------------------------------------------
__global__ __launch_bounds__(128)
void k_zreduce(const float* __restrict__ Zr_part, const float* __restrict__ Zw_part,
               const float* __restrict__ rv_part,
               float* __restrict__ rv_out, float* __restrict__ invZw)
{
  __shared__ float rr[64];
  __shared__ float rw[64];
  __shared__ float sInv;
  const int b = blockIdx.x;
  const int t = threadIdx.x;

  if (t < 64) {
    rr[t] = Zr_part[(size_t)t * BB + b];
    rw[t] = Zw_part[(size_t)t * BB + b];
  }
  __syncthreads();
  for (int s = 32; s > 0; s >>= 1) {
    if (t < s) { rr[t] += rr[t + s]; rw[t] += rw[t + s]; }
    __syncthreads();
  }
  if (t == 0) {
    sInv = 1.0f / rr[0];
    invZw[b] = 1.0f / rw[0];
  }
  __syncthreads();

  float s = 0.f;
#pragma unroll 8
  for (int c = 0; c < 64; ++c)
    s += rv_part[((size_t)c * BB + b) * DD + t];
  rv_out[(size_t)b * DD + t] = s * sInv;
}

// ---------------------------------------------------------------------------
// K5: normalize write_w in place and build new_memory.
//   em = (write_w_norm^T @ erase)/B ; am likewise ;
//   new_mem = mem*(1-em) + am.  grid = M/64 blocks, 8 waves.
// ---------------------------------------------------------------------------
__global__ __launch_bounds__(256)
void k_newmem(float* __restrict__ ww, const float* __restrict__ invZw,
              const bf16* __restrict__ eT, const bf16* __restrict__ aT,
              const float* __restrict__ mem, float* __restrict__ newmem)
{
  __shared__ float sInv[BB];
  __shared__ bf16 pT[64][40];        // [m][b] transposed normalized tile

  const int m0 = blockIdx.x * 64;
  const int t = threadIdx.x;
  const int w = t >> 5, lane = t & 31;
  const int mrw = w & 3, dbase = (w >> 2) * 64;
  const int lh = lane >> 4, ln = lane & 15;

  for (int i = t; i < BB; i += 256) sInv[i] = invZw[i];
  __syncthreads();

  v8f em[4], am[4];
#pragma unroll
  for (int n = 0; n < 4; ++n) {
    v8f z = {0.f, 0.f, 0.f, 0.f, 0.f, 0.f, 0.f, 0.f};
    em[n] = z; am[n] = z;
  }

  for (int b0 = 0; b0 < BB; b0 += 32) {
    // stage 32b x 64m tile: normalize write_w in place, transpose to LDS
    for (int idx = t; idx < 2048; idx += 256) {
      const int bb = idx >> 6, mm = idx & 63;
      const size_t off = (size_t)(b0 + bb) * MM + m0 + mm;
      const float nv = ww[off] * sInv[b0 + bb];
      ww[off] = nv;                                   // final write_w output
      pT[mm][bb] = (bf16)(nv * INV_B);
    }
    __syncthreads();

    AV ap;
    {
      const bf16* prow = &pT[mrw * 16 + ln][0];
      ap.h[0] = *(const v8bf*)(prow + lh * 16);
      ap.h[1] = *(const v8bf*)(prow + lh * 16 + 8);
    }
#pragma unroll
    for (int n = 0; n < 4; ++n) {
      const int d = dbase + n * 16 + ln;
      v16bf bve = *(const v16bf*)(eT + (size_t)d * BB + b0 + lh * 16);
      v16bf bva = *(const v16bf*)(aT + (size_t)d * BB + b0 + lh * 16);
      em[n] = wmma_bf16(ap.v, bve, em[n]);
      am[n] = wmma_bf16(ap.v, bva, am[n]);
    }
    __syncthreads();
  }

#pragma unroll
  for (int n = 0; n < 4; ++n) {
#pragma unroll
    for (int r = 0; r < 8; ++r) {
      const int m = m0 + mrw * 16 + r + 8 * lh;
      const int d = dbase + n * 16 + ln;
      const size_t o = (size_t)m * DD + d;
      newmem[o] = mem[o] * (1.0f - em[n][r]) + am[n][r];
    }
  }
}

// ---------------------------------------------------------------------------
extern "C" void kernel_launch(void* const* d_in, const int* in_sizes, int n_in,
                              void* d_out, int out_size, void* d_ws, size_t ws_size,
                              hipStream_t stream) {
  const float* h   = (const float*)d_in[0];
  const float* mem = (const float*)d_in[1];
  const float* Wr  = (const float*)d_in[2];
  const float* br  = (const float*)d_in[3];
  const float* Ww  = (const float*)d_in[4];
  const float* bw  = (const float*)d_in[5];
  const float* Wk  = (const float*)d_in[6];
  const float* bk  = (const float*)d_in[7];
  const float* We  = (const float*)d_in[8];
  const float* be  = (const float*)d_in[9];
  const float* Wa  = (const float*)d_in[10];
  const float* ba  = (const float*)d_in[11];

  float* out    = (float*)d_out;
  float* rv_out = out;                                   // (B,D)
  float* ww     = out + (size_t)BB * DD;                 // (B,M)
  float* nm     = ww + (size_t)BB * MM;                  // (M,D)

  char* ws = (char*)d_ws;
  size_t off = 0;
  bf16* qr   = (bf16*)(ws + off); off += (size_t)BB * DD * 2;       // 128K
  bf16* qw   = (bf16*)(ws + off); off += (size_t)BB * DD * 2;
  bf16* eT   = (bf16*)(ws + off); off += (size_t)DD * BB * 2;
  bf16* aT   = (bf16*)(ws + off); off += (size_t)DD * BB * 2;
  float* iZw = (float*)(ws + off); off += (size_t)BB * 4;
  off = (off + 255) & ~(size_t)255;
  float* Zrp = (float*)(ws + off); off += (size_t)64 * BB * 4;
  float* Zwp = (float*)(ws + off); off += (size_t)64 * BB * 4;
  float* rvp = (float*)(ws + off); off += (size_t)64 * BB * DD * 4; // 16MB
  bf16* mn   = (bf16*)(ws + off); off += (size_t)MM * DD * 2;       // 32MB
  bf16* mT   = (bf16*)(ws + off); off += (size_t)MM * DD * 2;       // 32MB

  k_proj<<<BB, 256, 0, stream>>>(h, Wr, br, Ww, bw, Wk, bk, We, be, Wa, ba,
                                 qr, qw, eT, aT);
  k_memprep<<<MM / 64, 256, 0, stream>>>(mem, mn, mT);
  dim3 g3(64, 8);
  k_attn<<<g3, 256, 0, stream>>>(qr, qw, mn, mT, ww, Zrp, Zwp, rvp);
  k_zreduce<<<BB, 128, 0, stream>>>(Zrp, Zwp, rvp, rv_out, iZw);
  k_newmem<<<MM / 64, 256, 0, stream>>>(ww, iZw, eT, aT, mem, nm);
}
